// NeuralModel_74268574482520
// MI455X (gfx1250) — matrix operand order repaired and checked by reading, not compiled, not run on hardware
//
#include <hip/hip_runtime.h>
#include <hip/hip_bf16.h>
#include <math.h>

typedef __attribute__((ext_vector_type(16))) __bf16 v16bf;
typedef __attribute__((ext_vector_type(8)))  float  v8f;

// ---------------------------------------------------------------- WMMA helpers

__device__ __forceinline__ v8f wmma_bf16(v16bf a, v16bf b, v8f c) {
  // D = A(16x32) x B(32x16) + C, f32 accumulate
  return __builtin_amdgcn_wmma_f32_16x16x32_bf16(false, a, false, b, (short)0, c,
                                                 false, false);
}

// A tile (16 rows x 32 K) from row-major [., ld] source (global or LDS).
// ISA layout: lane l (m = l&15, g = l>>4): elems 0..7 <-> K = k0+8g+e,
//             elems 8..15 <-> K = k0+16+8g+(e-8).
__device__ __forceinline__ v16bf ld_tileA(const __bf16* base, int ld, int m0, int k0) {
  int lane = threadIdx.x & 31;
  int m = m0 + (lane & 15);
  int g = lane >> 4;
  const __bf16* p = base + (size_t)m * ld + k0 + 8 * g;
  v16bf a;
  reinterpret_cast<uint4*>(&a)[0] = *reinterpret_cast<const uint4*>(p);
  reinterpret_cast<uint4*>(&a)[1] = *reinterpret_cast<const uint4*>(p + 16);
  return a;
}

// B tile (32 K x 16 N) built from rows n0..n0+15 of row-major W [N, ld] (B = W^T).
// ISA layout: lane l (n = l&15, g = l>>4): elems 0..7 <-> K = k0+8g+e, elems 8..15 <-> K = k0+16+8g+(e-8).
__device__ __forceinline__ v16bf ld_tileB(const __bf16* W, int ld, int n0, int k0) {
  int lane = threadIdx.x & 31;
  int n = n0 + (lane & 15);
  int g = lane >> 4;
  const __bf16* p = W + (size_t)n * ld + k0 + 8 * g;
  v16bf b;
  reinterpret_cast<uint4*>(&b)[0] = *reinterpret_cast<const uint4*>(p);
  reinterpret_cast<uint4*>(&b)[1] = *reinterpret_cast<const uint4*>(p + 16);
  return b;
}

__device__ __forceinline__ float sigf(float x) { return 1.0f / (1.0f + __expf(-x)); }

// ---------------------------------------------------------------- small utils

__global__ void k_f32_to_bf16(const float* __restrict__ s, __bf16* __restrict__ d, int n) {
  int i = blockIdx.x * 256 + threadIdx.x;
  if (i < n) d[i] = (__bf16)s[i];
}

// out[r,e] = embed[tok[r], e], E = 256, emitted as bf16
__global__ void k_gather_embed(const int* __restrict__ tok, const float* __restrict__ emb,
                               __bf16* __restrict__ out, int total) {
  int i = blockIdx.x * 256 + threadIdx.x;
  if (i < total) {
    int r = i >> 8;
    int e = i & 255;
    out[i] = (__bf16)emb[(size_t)tok[r] * 256 + e];
  }
}

// ---------------------------------------------------------------- C = A @ W^T (f32 out)
// A [M,K] bf16 row-major, W [N,K] bf16 row-major. One wave -> 16x64 tile.
// A-load for step k+32 is hoisted above the WMMA block for per-wave ILP.
__global__ __launch_bounds__(256) void k_gemm_bf16_nt(
    const __bf16* __restrict__ A, const __bf16* __restrict__ W,
    float* __restrict__ C, int M, int N, int K) {
  int tilesN = N >> 6;
  int id = blockIdx.x * 8 + (threadIdx.x >> 5);
  int total = (M >> 4) * tilesN;
  if (id >= total) return;
  int m0 = (id / tilesN) << 4;
  int n0 = (id % tilesN) << 6;
  v8f zero = {0.f, 0.f, 0.f, 0.f, 0.f, 0.f, 0.f, 0.f};
  v8f acc[4] = {zero, zero, zero, zero};
  v16bf a = ld_tileA(A, K, m0, 0);
  for (int k0 = 0; k0 < K; k0 += 32) {
    v16bf b0 = ld_tileB(W, K, n0 + 0, k0);
    v16bf b1 = ld_tileB(W, K, n0 + 16, k0);
    v16bf b2 = ld_tileB(W, K, n0 + 32, k0);
    v16bf b3 = ld_tileB(W, K, n0 + 48, k0);
    v16bf an = ld_tileA(A, K, m0, (k0 + 32 < K) ? (k0 + 32) : 0);
    acc[0] = wmma_bf16(a, b0, acc[0]);
    acc[1] = wmma_bf16(a, b1, acc[1]);
    acc[2] = wmma_bf16(a, b2, acc[2]);
    acc[3] = wmma_bf16(a, b3, acc[3]);
    a = an;
  }
  int lane = threadIdx.x & 31;
  int nl = lane & 15, mh = lane >> 4;
#pragma unroll
  for (int j = 0; j < 4; ++j)
#pragma unroll
    for (int r = 0; r < 8; ++r)
      C[(size_t)(m0 + mh * 8 + r) * N + (n0 + 16 * j + nl)] = acc[j][r];
}

// ---------------------------------------------------------------- LSTM scan
// One block = one direction (blockIdx.x = dir). 1024 threads = 32 waves.
// G [dir][T*32, 4*KO] f32 holds x@W_ih^T; Whh [dir][4*KO, KO] bf16; h kept in LDS (bf16).
// Each wave owns hidden slice jb..jb+KO/32, computing its i/f/g/o gate columns.
template <int KO>
__global__ __launch_bounds__(1024) void k_lstm_scan(
    const float* __restrict__ G, size_t gDirStride,
    const __bf16* __restrict__ Whh,
    const float* __restrict__ bih, const float* __restrict__ bhh,
    const float* __restrict__ h0, const float* __restrict__ c0,
    __bf16* __restrict__ ybf, int ybld,  // optional bf16 y [T*32, ybld] at col dir*KO
    float* __restrict__ hfin, float* __restrict__ cfin,  // optional [32, 2*KO] at col dir*KO
    int T, int revMask) {
  constexpr int NJT = KO / 512;
  constexpr int NG = 4 * KO;
  __shared__ __align__(16) __bf16 hsh[32 * KO];

  int dir = blockIdx.x;
  const float* Gd = G + (size_t)dir * gDirStride;
  const __bf16* Wd = Whh + (size_t)dir * NG * KO;
  const float* bid = bih + (size_t)dir * NG;
  const float* bhd = bhh + (size_t)dir * NG;
  int co = dir * KO;
  int rev = (revMask >> dir) & 1;
  int lane = threadIdx.x & 31;
  int wave = threadIdx.x >> 5;
  int jb = wave * (KO / 32);
  int nl = lane & 15, mh = lane >> 4;

  float creg[2][NJT][8];
#pragma unroll
  for (int mt = 0; mt < 2; ++mt)
#pragma unroll
    for (int jt = 0; jt < NJT; ++jt) {
      int j = jb + jt * 16 + nl;
#pragma unroll
      for (int r = 0; r < 8; ++r) {
        int m = mt * 16 + mh * 8 + r;
        float hv = h0 ? h0[(size_t)m * KO + j] : 0.0f;
        creg[mt][jt][r] = c0 ? c0[(size_t)m * KO + j] : 0.0f;
        hsh[m * KO + j] = (__bf16)hv;
      }
    }
  __syncthreads();

  v8f zero = {0.f, 0.f, 0.f, 0.f, 0.f, 0.f, 0.f, 0.f};
  for (int s = 0; s < T; ++s) {
    int t = rev ? (T - 1 - s) : s;
#pragma unroll
    for (int mt = 0; mt < 2; ++mt) {
      // pass over one batch half: reads only h rows [mt*16, mt*16+16)
      v8f acc[4][NJT];
#pragma unroll
      for (int gi = 0; gi < 4; ++gi)
#pragma unroll
        for (int jt = 0; jt < NJT; ++jt) acc[gi][jt] = zero;
      for (int kk = 0; kk < KO; kk += 32) {
        v16bf a = ld_tileA(hsh, KO, mt * 16, kk);
#pragma unroll
        for (int gi = 0; gi < 4; ++gi)
#pragma unroll
          for (int jt = 0; jt < NJT; ++jt) {
            v16bf b = ld_tileB(Wd, KO, gi * KO + jb + jt * 16, kk);
            acc[gi][jt] = wmma_bf16(a, b, acc[gi][jt]);
          }
      }
      __syncthreads();  // all waves done reading h rows of this half
#pragma unroll
      for (int jt = 0; jt < NJT; ++jt) {
        int j = jb + jt * 16 + nl;
        float bsum[4];
#pragma unroll
        for (int gi = 0; gi < 4; ++gi) bsum[gi] = bid[gi * KO + j] + bhd[gi * KO + j];
#pragma unroll
        for (int r = 0; r < 8; ++r) {
          int m = mt * 16 + mh * 8 + r;
          size_t grow = ((size_t)t * 32 + m) * NG;
          float gi_ = acc[0][jt][r] + Gd[grow + 0 * KO + j] + bsum[0];
          float gf_ = acc[1][jt][r] + Gd[grow + 1 * KO + j] + bsum[1];
          float gg_ = acc[2][jt][r] + Gd[grow + 2 * KO + j] + bsum[2];
          float go_ = acc[3][jt][r] + Gd[grow + 3 * KO + j] + bsum[3];
          float cn = sigf(gf_) * creg[mt][jt][r] + sigf(gi_) * tanhf(gg_);
          float hv = sigf(go_) * tanhf(cn);
          creg[mt][jt][r] = cn;
          hsh[m * KO + j] = (__bf16)hv;
          if (ybf) ybf[((size_t)t * 32 + m) * ybld + co + j] = (__bf16)hv;
          if ((s == T - 1) && hfin) {
            hfin[(size_t)m * (2 * KO) + co + j] = hv;
            cfin[(size_t)m * (2 * KO) + co + j] = cn;
          }
        }
      }
      __syncthreads();  // new h rows of this half visible before anyone re-reads
    }
  }
}

// ---------------------------------------------------------------- projection + online log-sum-exp
// Hid [2016,1024] bf16, Wp [32000,1024] bf16. Grid (126 row-tiles, 40 vocab splits of 800).
#define NSPLIT 40
#define VCHUNK 800

__global__ __launch_bounds__(256) void k_proj_lse(const __bf16* __restrict__ Hid,
                                                  const __bf16* __restrict__ Wp,
                                                  float2* __restrict__ part) {
  int rt = blockIdx.x;
  int sp = blockIdx.y;
  __shared__ __align__(16) __bf16 Ash[16 * 1024];
  __shared__ float mred[8][16];
  __shared__ float sred[8][16];
  // Stage the 16x1024 bf16 A tile into LDS with CDNA5 async copies (ASYNCcnt).
  {
    const __bf16* src = Hid + (size_t)rt * 16 * 1024;
    unsigned ldsbase = (unsigned)(uintptr_t)(&Ash[0]);
    for (int i = threadIdx.x; i < 2048; i += 256) {  // 2048 x 16B = 32KB
      unsigned lo = ldsbase + (unsigned)i * 16u;
      unsigned long long ga =
          (unsigned long long)(uintptr_t)src + (unsigned long long)i * 16ull;
      asm volatile("global_load_async_to_lds_b128 %0, %1, off" ::"v"(lo), "v"(ga)
                   : "memory");
    }
    asm volatile("s_wait_asynccnt 0" ::: "memory");
  }
  __syncthreads();

  int wave = threadIdx.x >> 5, lane = threadIdx.x & 31;
  float mrun[8], srun[8];
#pragma unroll
  for (int r = 0; r < 8; ++r) { mrun[r] = -1e30f; srun[r] = 0.0f; }
  v8f zero = {0.f, 0.f, 0.f, 0.f, 0.f, 0.f, 0.f, 0.f};
  for (int nt = wave; nt < VCHUNK / 16; nt += 8) {
    int n0 = sp * VCHUNK + nt * 16;
    // hint: prefetch the tile this wave touches next (W_proj streams from L2)
    if (nt + 8 < VCHUNK / 16) {
      const char* pf = (const char*)(Wp + ((size_t)sp * VCHUNK + (size_t)(nt + 8) * 16) * 1024);
      __builtin_prefetch(pf + (size_t)lane * 1024, 0, 1);
    }
    // K unrolled x2 with split accumulators to break the WMMA RAW chain
    v8f acc0 = zero, acc1 = zero;
    for (int k0 = 0; k0 < 1024; k0 += 64) {
      v16bf a0 = ld_tileA(Ash, 1024, 0, k0);
      v16bf b0 = ld_tileB(Wp, 1024, n0, k0);
      v16bf a1 = ld_tileA(Ash, 1024, 0, k0 + 32);
      v16bf b1 = ld_tileB(Wp, 1024, n0, k0 + 32);
      acc0 = wmma_bf16(a0, b0, acc0);
      acc1 = wmma_bf16(a1, b1, acc1);
    }
    v8f acc = acc0 + acc1;
#pragma unroll
    for (int r = 0; r < 8; ++r) {
      float v = acc[r];
      float tm = v;
      tm = fmaxf(tm, __shfl_xor(tm, 1));
      tm = fmaxf(tm, __shfl_xor(tm, 2));
      tm = fmaxf(tm, __shfl_xor(tm, 4));
      tm = fmaxf(tm, __shfl_xor(tm, 8));
      float te = __expf(v - tm);
      te += __shfl_xor(te, 1);
      te += __shfl_xor(te, 2);
      te += __shfl_xor(te, 4);
      te += __shfl_xor(te, 8);
      float mn = fmaxf(mrun[r], tm);
      srun[r] = srun[r] * __expf(mrun[r] - mn) + te * __expf(tm - mn);
      mrun[r] = mn;
    }
  }
  if ((lane & 15) == 0) {
    int h = lane >> 4;
#pragma unroll
    for (int r = 0; r < 8; ++r) {
      mred[wave][h * 8 + r] = mrun[r];
      sred[wave][h * 8 + r] = srun[r];
    }
  }
  __syncthreads();
  if (threadIdx.x < 16) {
    int rr = threadIdx.x;
    float m = -1e30f, s = 0.0f;
    for (int w = 0; w < 8; ++w) m = fmaxf(m, mred[w][rr]);
    for (int w = 0; w < 8; ++w) s += sred[w][rr] * __expf(mred[w][rr] - m);
    part[((size_t)rt * NSPLIT + sp) * 16 + rr] = make_float2(m, s);
  }
}

__global__ void k_logz(const float2* __restrict__ part, float* __restrict__ logZ) {
  int row = blockIdx.x * 256 + threadIdx.x;
  if (row >= 2016) return;
  int rt = row >> 4, rr = row & 15;
  float m = -1e30f, s = 0.0f;
  for (int sp = 0; sp < NSPLIT; ++sp) {
    float2 p = part[((size_t)rt * NSPLIT + sp) * 16 + rr];
    float mn = fmaxf(m, p.x);
    s = s * __expf(m - mn) + p.y * __expf(p.x - mn);
    m = mn;
  }
  logZ[row] = m + __logf(s);
}

// target logit per (t,b): dot(hidden[row], W_proj[tgt]) — one wave per row
__global__ __launch_bounds__(256) void k_tlogit(const __bf16* __restrict__ Hid,
                                                const __bf16* __restrict__ Wp,
                                                const int* __restrict__ hyps,
                                                float* __restrict__ tl) {
  int row = blockIdx.x * 8 + (threadIdx.x >> 5);
  if (row >= 2016) return;
  int lane = threadIdx.x & 31;
  int t = row >> 5, b = row & 31;
  int tgt = hyps[(t + 1) * 32 + b];
  const __bf16* hrow = Hid + (size_t)row * 1024;
  const __bf16* wrow = Wp + (size_t)tgt * 1024;
  float s = 0.0f;
  for (int k = lane; k < 1024; k += 32) s += (float)hrow[k] * (float)wrow[k];
  s += __shfl_xor(s, 1);
  s += __shfl_xor(s, 2);
  s += __shfl_xor(s, 4);
  s += __shfl_xor(s, 8);
  s += __shfl_xor(s, 16);
  if (lane == 0) tl[row] = s;
}

__global__ void k_scores(const float* __restrict__ tl, const float* __restrict__ logZ,
                         const int* __restrict__ hyps, float* __restrict__ out) {
  int b = threadIdx.x;
  if (b >= 32) return;
  float s = 0.0f;
  for (int t = 0; t < 63; ++t) {
    int tgt = hyps[(t + 1) * 32 + b];
    if (tgt != 0) s += tl[t * 32 + b] - logZ[t * 32 + b];
  }
  out[b] = s;
}

// ---------------------------------------------------------------- host driver

extern "C" void kernel_launch(void* const* d_in, const int* in_sizes, int n_in,
                              void* d_out, int out_size, void* d_ws, size_t ws_size,
                              hipStream_t stream) {
  (void)in_sizes; (void)n_in; (void)out_size; (void)ws_size;
  const int* prems = (const int*)d_in[0];    // [128,32]
  const int* hyps = (const int*)d_in[1];     // [64,32]
  const float* embed = (const float*)d_in[2];
  const float* Wih0 = (const float*)d_in[3];
  const float* Whh0 = (const float*)d_in[4];
  const float* bih0 = (const float*)d_in[5];
  const float* bhh0 = (const float*)d_in[6];
  const float* Wih1 = (const float*)d_in[7];
  const float* Whh1 = (const float*)d_in[8];
  const float* bih1 = (const float*)d_in[9];
  const float* bhh1 = (const float*)d_in[10];
  const float* Wdih = (const float*)d_in[11];
  const float* Wdhh = (const float*)d_in[12];
  const float* bdih = (const float*)d_in[13];
  const float* bdhh = (const float*)d_in[14];
  const float* Wproj = (const float*)d_in[15];
  float* out = (float*)d_out;

  char* p = (char*)d_ws;
  auto alloc = [&](size_t bytes) -> char* {
    char* r = p;
    p += (bytes + 255) & ~(size_t)255;
    return r;
  };
  __bf16* bWih0 = (__bf16*)alloc((size_t)2 * 2048 * 256 * 2);
  __bf16* bWhh0 = (__bf16*)alloc((size_t)2 * 2048 * 512 * 2);
  __bf16* bWih1 = (__bf16*)alloc((size_t)2 * 2048 * 1024 * 2);
  __bf16* bWhh1 = (__bf16*)alloc((size_t)2 * 2048 * 512 * 2);
  __bf16* bWdih = (__bf16*)alloc((size_t)4096 * 256 * 2);
  __bf16* bWdhh = (__bf16*)alloc((size_t)4096 * 1024 * 2);
  __bf16* bWproj = (__bf16*)alloc((size_t)32000 * 1024 * 2);
  __bf16* X0b = (__bf16*)alloc((size_t)4096 * 256 * 2);
  __bf16* Ydb = (__bf16*)alloc((size_t)2016 * 256 * 2);
  float* Gbuf = (float*)alloc((size_t)2 * 4096 * 2048 * 4);  // reused L0 / L1 / dec
  __bf16* y0b = (__bf16*)alloc((size_t)4096 * 1024 * 2);     // layer0 bf16 output (concat)
  float* h1 = (float*)alloc((size_t)32 * 1024 * 4);
  float* c1 = (float*)alloc((size_t)32 * 1024 * 4);
  __bf16* hidb = (__bf16*)alloc((size_t)2016 * 1024 * 2);
  float2* part = (float2*)alloc((size_t)126 * NSPLIT * 16 * 8);
  float* logZ = (float*)alloc((size_t)2016 * 4);
  float* tlog = (float*)alloc((size_t)2016 * 4);

  auto cvt = [&](const float* s, __bf16* d, int n) {
    k_f32_to_bf16<<<(n + 255) / 256, 256, 0, stream>>>(s, d, n);
  };
  // weight conversions
  cvt(Wih0, bWih0, 2 * 2048 * 256);
  cvt(Whh0, bWhh0, 2 * 2048 * 512);
  cvt(Wih1, bWih1, 2 * 2048 * 1024);
  cvt(Whh1, bWhh1, 2 * 2048 * 512);
  cvt(Wdih, bWdih, 4096 * 256);
  cvt(Wdhh, bWdhh, 4096 * 1024);
  cvt(Wproj, bWproj, 32000 * 1024);

  // embeddings
  k_gather_embed<<<(4096 * 256 + 255) / 256, 256, 0, stream>>>(prems, embed, X0b, 4096 * 256);
  k_gather_embed<<<(2016 * 256 + 255) / 256, 256, 0, stream>>>(hyps, embed, Ydb, 2016 * 256);

  auto gemm = [&](const __bf16* A, const __bf16* W, float* C, int M, int N, int K) {
    int tiles = (M / 16) * (N / 64);
    k_gemm_bf16_nt<<<(tiles + 7) / 8, 256, 0, stream>>>(A, W, C, M, N, K);
  };

  // layer 0: G0[dir] = X0 @ Wih0[dir]^T  -> scan (bf16 concat output straight to y0b)
  gemm(X0b, bWih0 + (size_t)0 * 2048 * 256, Gbuf + (size_t)0 * 4096 * 2048, 4096, 2048, 256);
  gemm(X0b, bWih0 + (size_t)1 * 2048 * 256, Gbuf + (size_t)1 * 4096 * 2048, 4096, 2048, 256);
  k_lstm_scan<512><<<2, 1024, 0, stream>>>(Gbuf, (size_t)4096 * 2048, bWhh0, bih0, bhh0,
                                           nullptr, nullptr, y0b, 1024, nullptr, nullptr,
                                           128, 2);

  // layer 1: G1[dir] = y0 @ Wih1[dir]^T -> scan (keep only final h/c)
  gemm(y0b, bWih1 + (size_t)0 * 2048 * 1024, Gbuf + (size_t)0 * 4096 * 2048, 4096, 2048, 1024);
  gemm(y0b, bWih1 + (size_t)1 * 2048 * 1024, Gbuf + (size_t)1 * 4096 * 2048, 4096, 2048, 1024);
  k_lstm_scan<512><<<2, 1024, 0, stream>>>(Gbuf, (size_t)4096 * 2048, bWhh1, bih1, bhh1,
                                           nullptr, nullptr, nullptr, 1024, h1, c1,
                                           128, 2);

  // decoder: Gdec = Y @ Wdih^T -> scan from (h1,c1), emit bf16 hidden
  gemm(Ydb, bWdih, Gbuf, 2016, 4096, 256);
  k_lstm_scan<1024><<<1, 1024, 0, stream>>>(Gbuf, 0, bWdhh, bdih, bdhh,
                                            h1, c1, hidb, 1024, nullptr, nullptr,
                                            63, 0);

  // fused projection + log-softmax
  dim3 pg(126, NSPLIT);
  k_proj_lse<<<pg, 256, 0, stream>>>(hidb, bWproj, part);
  k_logz<<<(2016 + 255) / 256, 256, 0, stream>>>(part, logZ);
  k_tlogit<<<(2016 + 7) / 8, 256, 0, stream>>>(hidb, bWproj, hyps, tlog);
  k_scores<<<1, 32, 0, stream>>>(tlog, logZ, hyps, out);
}
